// LSTM2d_46488726012659
// MI455X (gfx1250) — matrix operand order repaired and checked by reading, not compiled, not run on hardware
//
#include <hip/hip_runtime.h>
#include <hip/hip_bf16.h>

// ---------------------------------------------------------------------------
// Problem constants (from reference): V=32000, E=256, HE=256, S=256, L=48, B=32
// ---------------------------------------------------------------------------
#define CV   32000
#define CE   256
#define CHE  256
#define CS   256
#define CL   48
#define CB   32
#define CLB  (CL * CB)          // 1536 rows for (l,b)-flattened GEMMs

typedef __attribute__((ext_vector_type(16))) __bf16 v16bf;
typedef __attribute__((ext_vector_type(8)))  __bf16 v8bf;
typedef __attribute__((ext_vector_type(8)))  float  v8f;

__device__ inline float sigmoidf_(float x) { return 1.0f / (1.0f + __expf(-x)); }

// ---------------------------------------------------------------------------
// Wave-level 16x16 bf16 WMMA tile:  acc += A(16xK, row-major, lda) * W(16xK)^T
//
// Per-lane fragment slices are contiguous, so each operand is 2 x b128 loads:
//   A (16-bit 16x32, ISA 7.12.2): lane(m,h) -> k in [h*8,h*8+8) and +16
//   B (32x16 from row-major W):   lane(n,h) -> k in [h*8,h*8+8) and +16
// lda/ldw must be multiples of 8 elements; bases 16B-aligned (true here).
// ---------------------------------------------------------------------------
template <typename AP, typename BP>
__device__ inline v8f wmma_bf16_acc(AP A, int lda, BP W, int ldw, int K, v8f acc) {
  const int lane = threadIdx.x & 31;
  const int m    = lane & 15;
  const int hh   = lane >> 4;
  AP arow = A + m * lda + hh * 8;     // chunks at +0 and +16 (elements)
  BP wrow = W + m * ldw + hh * 8;     // chunks at +0 and +16
  for (int k0 = 0; k0 < K; k0 += 32) {
    __builtin_prefetch((const void*)(wrow + k0 + 128), 0, 1);
    union { v16bf v; v8bf h[2]; } a, b;
    a.h[0] = *(const v8bf*)(arow + k0);
    a.h[1] = *(const v8bf*)(arow + k0 + 16);
    b.h[0] = *(const v8bf*)(wrow + k0);
    b.h[1] = *(const v8bf*)(wrow + k0 + 16);
    acc = __builtin_amdgcn_wmma_f32_16x16x32_bf16(false, a.v, false, b.v,
                                                  (short)0, acc, false, false);
  }
  return acc;
}

// ---------------------------------------------------------------------------
// Elementwise f32 -> bf16 conversion
// ---------------------------------------------------------------------------
__global__ void f32_to_bf16_kernel(const float* __restrict__ src,
                                   __bf16* __restrict__ dst, int n) {
  int i = blockIdx.x * blockDim.x + threadIdx.x;
  if (i < n) dst[i] = (__bf16)src[i];
}

// ---------------------------------------------------------------------------
// Embedding gather (f32 table -> bf16 rows)
// ---------------------------------------------------------------------------
__global__ void gather_emb_kernel(const int* __restrict__ tok,
                                  const float* __restrict__ emb,
                                  __bf16* __restrict__ out) {
  const int row = blockIdx.x;           // l*B + b
  const int t   = tok[row];
  for (int e = threadIdx.x; e < CE; e += blockDim.x)
    out[(size_t)row * CE + e] = (__bf16)emb[(size_t)t * CE + e];
}

// ---------------------------------------------------------------------------
// Generic GEMM:  C[M,N] = A[M,K](bf16) * W[N,K](bf16)^T + bias[N], f32 out.
// block = 256 threads = 8 waves; wave -> one 16x16 tile.
// grid.x covers N/128 tiles-of-8, grid.y = M/16.
// ---------------------------------------------------------------------------
__global__ __launch_bounds__(256)
void gemm_bf16_bias_kernel(const __bf16* __restrict__ A,
                           const __bf16* __restrict__ W,
                           const float* __restrict__ bias,
                           float* __restrict__ C, int M, int N, int K) {
  const int wave = threadIdx.x >> 5;
  const int nt   = blockIdx.x * (blockDim.x >> 5) + wave;
  const int mt   = blockIdx.y;
  if (nt * 16 >= N) return;             // wave-uniform
  const int lane = threadIdx.x & 31;
  const int col  = nt * 16 + (lane & 15);
  const float bv = bias[col];
  v8f acc = {bv, bv, bv, bv, bv, bv, bv, bv};
  acc = wmma_bf16_acc(A + (size_t)mt * 16 * K, K,
                      W + (size_t)nt * 16 * K, K, K, acc);
  const int rowbase = mt * 16 + ((lane >> 4) << 3);
#pragma unroll
  for (int r = 0; r < 8; ++r)
    C[(size_t)(rowbase + r) * N + col] = acc[r];
}

// ---------------------------------------------------------------------------
// Bidirectional LSTM recurrence. grid = 2 (dir), block = 1024 (32 waves).
// h (bf16) and c (f32) live in LDS; each wave owns (m-tile, s-tile) and the
// 4 gate tiles for that slice, so the nonlinearity is lane-local.
// ---------------------------------------------------------------------------
__global__ __launch_bounds__(1024)
void bilstm_kernel(const float* __restrict__ preX_f,
                   const float* __restrict__ preX_b,
                   const __bf16* __restrict__ WhhF,
                   const __bf16* __restrict__ WhhB,
                   __bf16* __restrict__ hAll /* (L,B,2HE) bf16 */) {
  const int dir = blockIdx.x;
  const float*  preX = dir ? preX_b : preX_f;
  const __bf16* Whh  = dir ? WhhB : WhhF;

  __shared__ __bf16 hS[CB][CHE];   // 16 KB
  __shared__ float  cS[CB][CHE];   // 32 KB
  for (int i = threadIdx.x; i < CB * CHE; i += blockDim.x) {
    ((__bf16*)hS)[i] = (__bf16)0.0f;
    ((float*)cS)[i]  = 0.0f;
  }
  __syncthreads();

  const int lane = threadIdx.x & 31;
  const int wave = threadIdx.x >> 5;
  const int mt   = wave >> 4;            // 0..1  -> batch tile
  const int st   = wave & 15;            // 0..15 -> state tile
  const int ss   = st * 16 + (lane & 15);
  const int bb   = mt * 16 + ((lane >> 4) << 3);

  for (int step = 0; step < CL; ++step) {
    const int t = dir ? (CL - 1 - step) : step;
    v8f acc[4];
#pragma unroll
    for (int g = 0; g < 4; ++g) {
      const int n = g * CHE + ss;
#pragma unroll
      for (int r = 0; r < 8; ++r)
        acc[g][r] = preX[((size_t)t * CB + (bb + r)) * (4 * CHE) + n];
      acc[g] = wmma_bf16_acc(&hS[mt * 16][0], CHE,
                             Whh + (size_t)(g * CHE + st * 16) * CHE, CHE,
                             CHE, acc[g]);
    }
    __syncthreads();                     // all reads of hS done
#pragma unroll
    for (int r = 0; r < 8; ++r) {
      const int b = bb + r;
      const float iv = sigmoidf_(acc[0][r]);
      const float fv = sigmoidf_(acc[1][r]);
      const float gv = tanhf(acc[2][r]);
      const float ov = sigmoidf_(acc[3][r]);
      const float c  = fv * cS[b][ss] + iv * gv;
      cS[b][ss] = c;
      const float h = ov * tanhf(c);
      hS[b][ss] = (__bf16)h;
      hAll[((size_t)t * CB + b) * (2 * CHE) + dir * CHE + ss] = (__bf16)h;
    }
    __syncthreads();                     // writes visible for next step
  }
}

// ---------------------------------------------------------------------------
// One anti-diagonal of the 2D LSTM. grid.x = L (one block per row i),
// block = 1024 (32 waves). Double-buffered s/c; inactive rows copy through.
// Cell (i, j=d-i):  pre = [h[j], y_emb[i]] @ Wx^T + [s_hor, s_ver] @ Ws^T + b
// ---------------------------------------------------------------------------
__global__ __launch_bounds__(1024)
void lstm2d_diag_kernel(int d,
                        const __bf16* __restrict__ hAll,
                        const __bf16* __restrict__ yEmb,
                        const __bf16* __restrict__ Wx,
                        const __bf16* __restrict__ Ws,
                        const float* __restrict__ bc,
                        const float* __restrict__ sIn,
                        const float* __restrict__ cIn,
                        float* __restrict__ sOut,
                        float* __restrict__ cOut,
                        float* __restrict__ states) {
  const int i = blockIdx.x;
  const int j = d - i;
  if (j < 0 || j >= CL) {               // inactive row: carry state through
    const size_t base = (size_t)i * CB * CS;
    const uint4* si4 = (const uint4*)(sIn + base);
    const uint4* ci4 = (const uint4*)(cIn + base);
    uint4* so4 = (uint4*)(sOut + base);
    uint4* co4 = (uint4*)(cOut + base);
    for (int k = threadIdx.x; k < (CB * CS) / 4; k += blockDim.x) {
      so4[k] = si4[k];
      co4[k] = ci4[k];
    }
    return;
  }

  __shared__ __bf16 xcat[CB][2 * CHE + CE]; // 32x768 bf16 = 48 KB
  __shared__ __bf16 scat[CB][2 * CS];       // 32x512 bf16 = 32 KB

  // h[j] -> xcat[:, 0:512)  (bf16 copy, 16B chunks)
  for (int k = threadIdx.x; k < CB * 64; k += blockDim.x) {
    const int b = k >> 6, c8 = (k & 63) * 8;
    *(uint4*)&xcat[b][c8] =
        *(const uint4*)&hAll[((size_t)j * CB + b) * (2 * CHE) + c8];
  }
  // y_emb[i] -> xcat[:, 512:768)  (bf16 copy, 16B chunks)
  for (int k = threadIdx.x; k < CB * 32; k += blockDim.x) {
    const int b = k >> 5, c8 = (k & 31) * 8;
    *(uint4*)&xcat[b][2 * CHE + c8] =
        *(const uint4*)&yEmb[((size_t)i * CB + b) * CE + c8];
  }
  // s_hor / s_ver -> scat (f32 -> bf16)
  for (int k = threadIdx.x; k < CB * CS; k += blockDim.x) {
    const int b = k >> 8, col = k & 255;
    scat[b][col] = (__bf16)sIn[((size_t)i * CB + b) * CS + col];   // s_hor
    const float sv = (i > 0) ? sIn[((size_t)(i - 1) * CB + b) * CS + col]
                             : 0.0f;                               // s_ver
    scat[b][CS + col] = (__bf16)sv;
  }
  __syncthreads();

  const int lane = threadIdx.x & 31;
  const int wave = threadIdx.x >> 5;
  const int mt   = wave >> 4;
  const int st   = wave & 15;
  const int ss   = st * 16 + (lane & 15);
  const int bb   = mt * 16 + ((lane >> 4) << 3);

  v8f acc[5];
#pragma unroll
  for (int g = 0; g < 5; ++g) {
    const int n = g * CS + ss;
    const float bv = bc[n];
    acc[g] = (v8f){bv, bv, bv, bv, bv, bv, bv, bv};
    acc[g] = wmma_bf16_acc(&xcat[mt * 16][0], 2 * CHE + CE,
                           Wx + (size_t)(g * CS + st * 16) * (2 * CHE + CE),
                           2 * CHE + CE, 2 * CHE + CE, acc[g]);
    acc[g] = wmma_bf16_acc(&scat[mt * 16][0], 2 * CS,
                           Ws + (size_t)(g * CS + st * 16) * (2 * CS),
                           2 * CS, 2 * CS, acc[g]);
  }

  // gate order per reference: ig, fg, og, lg, gg
#pragma unroll
  for (int r = 0; r < 8; ++r) {
    const int b = bb + r;
    const float ig = sigmoidf_(acc[0][r]);
    const float fg = sigmoidf_(acc[1][r]);
    const float og = sigmoidf_(acc[2][r]);
    const float lg = sigmoidf_(acc[3][r]);
    const float gg = tanhf(acc[4][r]);
    const size_t idx = ((size_t)i * CB + b) * CS + ss;
    const float ch = cIn[idx];
    const float cv = (i > 0) ? cIn[((size_t)(i - 1) * CB + b) * CS + ss] : 0.0f;
    const float cn = fg * (lg * ch + (1.0f - lg) * cv) + ig * gg;
    const float sn = og * tanhf(cn);
    cOut[idx]   = cn;
    sOut[idx]   = sn;
    states[idx] = sn;   // last write (j == L-1) is the row's final state
  }
}

// ---------------------------------------------------------------------------
// In-place row softmax over V columns. One block per (l,b) row.
// ---------------------------------------------------------------------------
__global__ __launch_bounds__(256)
void softmax_kernel(float* __restrict__ out, int cols) {
  float* p = out + (size_t)blockIdx.x * cols;
  __shared__ float red[256];
  float mx = -3.4e38f;
  for (int c = threadIdx.x; c < cols; c += 256) mx = fmaxf(mx, p[c]);
  red[threadIdx.x] = mx;
  __syncthreads();
  for (int off = 128; off > 0; off >>= 1) {
    if (threadIdx.x < off)
      red[threadIdx.x] = fmaxf(red[threadIdx.x], red[threadIdx.x + off]);
    __syncthreads();
  }
  mx = red[0];
  __syncthreads();
  float sum = 0.0f;
  for (int c = threadIdx.x; c < cols; c += 256) {
    const float e = __expf(p[c] - mx);
    p[c] = e;
    sum += e;
  }
  red[threadIdx.x] = sum;
  __syncthreads();
  for (int off = 128; off > 0; off >>= 1) {
    if (threadIdx.x < off) red[threadIdx.x] += red[threadIdx.x + off];
    __syncthreads();
  }
  const float inv = 1.0f / red[0];
  for (int c = threadIdx.x; c < cols; c += 256) p[c] *= inv;
}

// ---------------------------------------------------------------------------
// Host-side orchestration
// ---------------------------------------------------------------------------
extern "C" void kernel_launch(void* const* d_in, const int* in_sizes, int n_in,
                              void* d_out, int out_size, void* d_ws,
                              size_t ws_size, hipStream_t stream) {
  (void)in_sizes; (void)n_in; (void)out_size; (void)ws_size;

  const int*   x        = (const int*)d_in[0];
  const int*   y        = (const int*)d_in[1];
  const float* emb      = (const float*)d_in[2];
  const float* Wih_f    = (const float*)d_in[3];
  const float* Whh_f    = (const float*)d_in[4];
  const float* b_f      = (const float*)d_in[5];
  const float* Wih_b    = (const float*)d_in[6];
  const float* Whh_b    = (const float*)d_in[7];
  const float* b_b      = (const float*)d_in[8];
  const float* Wx       = (const float*)d_in[9];
  const float* Ws       = (const float*)d_in[10];
  const float* b_cell   = (const float*)d_in[11];
  const float* W_logits = (const float*)d_in[12];
  const float* b_logits = (const float*)d_in[13];
  float* out = (float*)d_out;

  // ---- workspace carve-up (~44 MB total) ----
  char* ws = (char*)d_ws;
  auto alloc = [&](size_t bytes) -> char* {
    char* p = ws;
    ws += (bytes + 255) & ~(size_t)255;
    return p;
  };
  const int XW = 2 * CHE + CE;   // 768
  const int SW = 2 * CS;         // 512
  __bf16* xembB   = (__bf16*)alloc((size_t)CLB * CE * 2);
  __bf16* yembB   = (__bf16*)alloc((size_t)CLB * CE * 2);
  __bf16* WihFB   = (__bf16*)alloc((size_t)4 * CHE * CE * 2);
  __bf16* WhhFB   = (__bf16*)alloc((size_t)4 * CHE * CHE * 2);
  __bf16* WihBB   = (__bf16*)alloc((size_t)4 * CHE * CE * 2);
  __bf16* WhhBB   = (__bf16*)alloc((size_t)4 * CHE * CHE * 2);
  __bf16* WxB     = (__bf16*)alloc((size_t)5 * CS * XW * 2);
  __bf16* WsB     = (__bf16*)alloc((size_t)5 * CS * SW * 2);
  __bf16* WlogB   = (__bf16*)alloc((size_t)CV * CS * 2);
  __bf16* hAllB   = (__bf16*)alloc((size_t)CLB * 2 * CHE * 2);
  __bf16* statesB = (__bf16*)alloc((size_t)CLB * CS * 2);
  float*  preXf   = (float*)alloc((size_t)CLB * 4 * CHE * 4);
  float*  preXb   = (float*)alloc((size_t)CLB * 4 * CHE * 4);
  float*  sBuf0   = (float*)alloc((size_t)CLB * CS * 4);
  float*  sBuf1   = (float*)alloc((size_t)CLB * CS * 4);
  float*  cBuf0   = (float*)alloc((size_t)CLB * CS * 4);
  float*  cBuf1   = (float*)alloc((size_t)CLB * CS * 4);
  float*  states  = (float*)alloc((size_t)CLB * CS * 4);

  auto cvt = [&](const float* s, __bf16* dcv, int n) {
    f32_to_bf16_kernel<<<(n + 255) / 256, 256, 0, stream>>>(s, dcv, n);
  };

  // 1) weights -> bf16
  cvt(Wih_f, WihFB, 4 * CHE * CE);
  cvt(Whh_f, WhhFB, 4 * CHE * CHE);
  cvt(Wih_b, WihBB, 4 * CHE * CE);
  cvt(Whh_b, WhhBB, 4 * CHE * CHE);
  cvt(Wx, WxB, 5 * CS * XW);
  cvt(Ws, WsB, 5 * CS * SW);
  cvt(W_logits, WlogB, CV * CS);

  // 2) embedding gathers
  gather_emb_kernel<<<CLB, 256, 0, stream>>>(x, emb, xembB);
  gather_emb_kernel<<<CLB, 256, 0, stream>>>(y, emb, yembB);

  // 3) pre_x GEMMs: (1536 x 1024 x 256)
  {
    dim3 g((4 * CHE) / 128, CLB / 16);
    gemm_bf16_bias_kernel<<<g, 256, 0, stream>>>(xembB, WihFB, b_f, preXf,
                                                 CLB, 4 * CHE, CE);
    gemm_bf16_bias_kernel<<<g, 256, 0, stream>>>(xembB, WihBB, b_b, preXb,
                                                 CLB, 4 * CHE, CE);
  }

  // 4) bidirectional recurrence
  bilstm_kernel<<<2, 1024, 0, stream>>>(preXf, preXb, WhhFB, WhhBB, hAllB);

  // 5) 2D LSTM over 2L-1 anti-diagonals (double-buffered s/c)
  hipMemsetAsync(sBuf0, 0, (size_t)CLB * CS * 4, stream);
  hipMemsetAsync(cBuf0, 0, (size_t)CLB * CS * 4, stream);
  for (int d = 0; d < 2 * CL - 1; ++d) {
    const float* si = (d & 1) ? sBuf1 : sBuf0;
    const float* ci = (d & 1) ? cBuf1 : cBuf0;
    float* so = (d & 1) ? sBuf0 : sBuf1;
    float* co = (d & 1) ? cBuf0 : cBuf1;
    lstm2d_diag_kernel<<<CL, 1024, 0, stream>>>(d, hAllB, yembB, WxB, WsB,
                                                b_cell, si, ci, so, co, states);
  }

  // 6) logits GEMM (1536 x 32000 x 256) into d_out, then softmax in place
  cvt(states, statesB, CLB * CS);
  {
    dim3 g(CV / 128, CLB / 16);
    gemm_bf16_bias_kernel<<<g, 256, 0, stream>>>(statesB, WlogB, b_logits, out,
                                                 CLB, CV, CS);
  }
  softmax_kernel<<<CLB, 256, 0, stream>>>(out, CV);
}
